// MRISliceEncoder_66288525247126
// MI455X (gfx1250) — matrix operand order repaired and checked by reading, not compiled, not run on hardware
//
#include <hip/hip_runtime.h>

// ---------- types ----------
typedef unsigned short u16;
typedef __bf16 bf16_t;
typedef bf16_t v16bf __attribute__((ext_vector_type(16)));
typedef float  v8f_t __attribute__((ext_vector_type(8)));
typedef u16    v8u   __attribute__((ext_vector_type(8)));
typedef u16    v16u  __attribute__((ext_vector_type(16)));

#define N_SLICES 8000
#define IN_DIM   512
#define E_DIM    512
#define DI       1024
#define S_DIM    16
#define DC       4
#define DT_RANK  32
#define NL       2
#define CHUNK    500
#define NCHUNK   (N_SLICES / CHUNK)

// CDNA5 async global->LDS staging (ASYNCcnt path). Set to 0 to fall back to
// plain load + ds_store if the assembler rejects the mnemonic.
#define USE_ASYNC_LDS 1

__device__ __forceinline__ u16 f2bf(float f) {
    unsigned u = __builtin_bit_cast(unsigned, f);
    unsigned r = u + 0x7FFFu + ((u >> 16) & 1u);   // round-to-nearest-even
    return (u16)(r >> 16);
}

// copy 16 bytes global -> LDS
__device__ __forceinline__ void copy16_g2l(u16* lds, const u16* g) {
#if USE_ASYNC_LDS
    asm volatile("global_load_async_to_lds_b128 %0, %1, off"
                 :: "v"((unsigned)(unsigned long long)lds),
                    "v"((unsigned long long)g)
                 : "memory");
#else
    *(v8u*)lds = *(const v8u*)g;
#endif
}

__device__ __forceinline__ void wait_async() {
#if USE_ASYNC_LDS
    asm volatile("s_wait_asynccnt 0" ::: "memory");
#endif
}

// ---------- f32 -> bf16 converters ----------
__global__ __launch_bounds__(256) void cvt_f32_bf16(const float* __restrict__ src,
                                                    u16* __restrict__ dst, int n) {
    int gid = blockIdx.x * 256 + threadIdx.x;
    if (gid < n) dst[gid] = f2bf(src[gid]);
}

__global__ __launch_bounds__(256) void cvt2d_f32_bf16(const float* __restrict__ src,
                                                      u16* __restrict__ dst,
                                                      int rows, int scols, int cols) {
    int gid = blockIdx.x * 256 + threadIdx.x;
    if (gid < rows * cols) {
        int r = gid / cols, c = gid - r * cols;
        dst[gid] = f2bf(src[(size_t)r * scols + c]);
    }
}

// =====================================================================
// LDS-staged double-buffered WMMA GEMM:  C[M,N] = A[M,K]*B[N,K]^T (+bias)
// block = 256 threads (8 waves); block tile 64(M) x 256(N); K-step 32.
// waves: 2(M) x 4(N); each wave: 2x4 = eight 16x16 f32 accumulators.
// grid = (N/256, M/64). Requires M%64==0, N%256==0, K%32==0.
// =====================================================================
__global__ __launch_bounds__(256) void gemm_bf16_wmma_lds(const u16* __restrict__ A,
                                                          const u16* __restrict__ B,
                                                          const float* __restrict__ bias,
                                                          float* __restrict__ C,
                                                          int M, int N, int K) {
    __shared__ u16 sA[2][64 * 32];    //  4 KB per buffer
    __shared__ u16 sB[2][256 * 32];   // 16 KB per buffer

    const int tid  = threadIdx.x;
    const int lane = tid & 31;
    const int wave = tid >> 5;
    const int half = lane >> 4;
    const int l16  = lane & 15;
    const int wr   = wave >> 2;       // 0..1 : M sub-tile
    const int wc   = wave & 3;        // 0..3 : N sub-tile
    const int row0 = blockIdx.y * 64;
    const int col0 = blockIdx.x * 256;

    v8f_t acc[2][4];
#pragma unroll
    for (int mt = 0; mt < 2; mt++)
#pragma unroll
        for (int t = 0; t < 4; t++)
#pragma unroll
            for (int i = 0; i < 8; i++) acc[mt][t][i] = 0.0f;

    const int nk = K >> 5;   // number of 32-wide K tiles

    // cooperative stage of K-tile kt into buffer buf
    auto stage = [&](int kt, int buf) {
        const int k0 = kt << 5;
        {   // A: 64 rows x 32 -> 256 16B chunks, one per thread
            int r = tid >> 2, p = (tid & 3) << 3;
            copy16_g2l(&sA[buf][r * 32 + p], A + (size_t)(row0 + r) * K + k0 + p);
        }
#pragma unroll
        for (int i = 0; i < 4; i++) {  // B: 256 rows x 32 -> 1024 chunks
            int c = tid + (i << 8);
            int r = c >> 2, p = (c & 3) << 3;
            copy16_g2l(&sB[buf][r * 32 + p], B + (size_t)(col0 + r) * K + k0 + p);
        }
    };

    stage(0, 0);
    wait_async();
    __syncthreads();

    for (int kt = 0; kt < nk; kt++) {
        const int cur = kt & 1;
        if (kt + 1 < nk) stage(kt + 1, cur ^ 1);   // overlap DMA with WMMA

        // A fragments (per wave, 2 row tiles)
        v16bf afrag[2];
#pragma unroll
        for (int mt = 0; mt < 2; mt++) {
            const u16* ar = &sA[cur][(wr * 32 + mt * 16 + l16) * 32];
            afrag[mt] = __builtin_bit_cast(v16bf, *(const v16u*)(ar + 16 * half));
        }
        // B fragments (4 col tiles)
        v16bf bfrag[4];
#pragma unroll
        for (int t = 0; t < 4; t++) {
            const u16* br = &sB[cur][(wc * 64 + t * 16 + l16) * 32];
            bfrag[t] = __builtin_bit_cast(v16bf, *(const v16u*)(br + 16 * half));
        }
        // 8 WMMAs, A fragment reused across the N tiles
#pragma unroll
        for (int mt = 0; mt < 2; mt++)
#pragma unroll
            for (int t = 0; t < 4; t++)
                acc[mt][t] = __builtin_amdgcn_wmma_f32_16x16x32_bf16(
                    false, afrag[mt], false, bfrag[t], (short)0, acc[mt][t],
                    false, false);

        wait_async();      // staged tile fully in LDS before the barrier
        __syncthreads();
    }

#pragma unroll
    for (int mt = 0; mt < 2; mt++)
#pragma unroll
        for (int t = 0; t < 4; t++) {
            int col = col0 + wc * 64 + t * 16 + l16;
            float bs = bias ? bias[col] : 0.0f;
#pragma unroll
            for (int r = 0; r < 8; r++) {
                int row = row0 + wr * 32 + mt * 16 + half * 8 + r;
                C[(size_t)row * N + col] = acc[mt][t][r] + bs;
            }
        }
}

// ---------- direct (non-LDS) WMMA GEMM for small N (x-proj, N=64) ----------
template <int NT>
__global__ __launch_bounds__(128) void gemm_bf16_wmma_direct(const u16* __restrict__ A,
                                                             const u16* __restrict__ B,
                                                             const float* __restrict__ bias,
                                                             float* __restrict__ C,
                                                             int M, int N, int K) {
    const int lane = threadIdx.x & 31;
    const int wave = threadIdx.x >> 5;
    const int half = lane >> 4;
    const int l16  = lane & 15;
    const int row0 = blockIdx.y * 16;
    const int col0 = (blockIdx.x * 4 + wave) * (16 * NT);

    v8f_t acc[NT];
#pragma unroll
    for (int t = 0; t < NT; t++)
#pragma unroll
        for (int i = 0; i < 8; i++) acc[t][i] = 0.0f;

    const u16* arow = A + (size_t)(row0 + l16) * K;
    for (int k0 = 0; k0 < K; k0 += 32) {
        v16bf afrag = __builtin_bit_cast(v16bf, *(const v16u*)(arow + k0 + 16 * half));
#pragma unroll
        for (int t = 0; t < NT; t++) {
            const u16* brow = B + (size_t)(col0 + t * 16 + l16) * K;
            v16bf bfrag = __builtin_bit_cast(v16bf, *(const v16u*)(brow + k0 + 16 * half));
            acc[t] = __builtin_amdgcn_wmma_f32_16x16x32_bf16(
                false, afrag, false, bfrag, (short)0, acc[t], false, false);
        }
    }
#pragma unroll
    for (int t = 0; t < NT; t++) {
        int col = col0 + t * 16 + l16;
        float bs = bias ? bias[col] : 0.0f;
#pragma unroll
        for (int r = 0; r < 8; r++)
            C[(size_t)(row0 + half * 8 + r) * N + col] = acc[t][r] + bs;
    }
}

// ---------- LayerNorm over E=512 (optional residual); writes f32 + bf16 ----------
__global__ __launch_bounds__(256) void ln_kernel(const float* __restrict__ in,
                                                 const float* __restrict__ res,
                                                 const float* __restrict__ g,
                                                 const float* __restrict__ b,
                                                 float* __restrict__ outf,
                                                 u16* __restrict__ outb) {
    __shared__ float s1[256], s2[256];
    const int tid = threadIdx.x;
    const size_t base = (size_t)blockIdx.x * E_DIM;
    float v0 = in[base + tid];
    float v1 = in[base + tid + 256];
    if (res) { v0 += res[base + tid]; v1 += res[base + tid + 256]; }
    s1[tid] = v0 + v1;
    s2[tid] = v0 * v0 + v1 * v1;
    __syncthreads();
    for (int off = 128; off > 0; off >>= 1) {
        if (tid < off) { s1[tid] += s1[tid + off]; s2[tid] += s2[tid + off]; }
        __syncthreads();
    }
    float mu = s1[0] * (1.0f / E_DIM);
    float var = s2[0] * (1.0f / E_DIM) - mu * mu;
    float rs = rsqrtf(var + 1e-5f);
    float o0 = (v0 - mu) * rs * g[tid] + b[tid];
    float o1 = (v1 - mu) * rs * g[tid + 256] + b[tid + 256];
    outf[base + tid] = o0;
    outf[base + tid + 256] = o1;
    outb[base + tid] = f2bf(o0);
    outb[base + tid + 256] = f2bf(o1);
}

// ---------- depthwise causal conv(4) + bias + SiLU on x half of xz ----------
__global__ __launch_bounds__(256) void conv_silu_kernel(const float* __restrict__ xz,
                                                        const float* __restrict__ w,
                                                        const float* __restrict__ b,
                                                        float* __restrict__ xf,
                                                        u16* __restrict__ xb) {
    int gid = blockIdx.x * 256 + threadIdx.x;
    if (gid >= N_SLICES * DI) return;
    int d = gid & (DI - 1);
    int t = gid >> 10;
    int l = t % CHUNK;
    int tb = t - l;
    float acc = b[d];
    const float* wd = w + d * DC;
#pragma unroll
    for (int j = 0; j < DC; j++) {
        int li = l - (DC - 1) + j;
        if (li >= 0) acc += xz[(size_t)(tb + li) * (2 * DI) + d] * wd[j];
    }
    float s = acc * (1.0f / (1.0f + __expf(-acc)));
    xf[gid] = s;
    xb[gid] = f2bf(s);
}

// ---------- dt = softplus(dt_raw + dt_b) in place ----------
__global__ __launch_bounds__(256) void softplus_kernel(float* __restrict__ dt,
                                                       const float* __restrict__ bvec) {
    int gid = blockIdx.x * 256 + threadIdx.x;
    if (gid >= N_SLICES * DI) return;
    int d = gid & (DI - 1);
    float x = dt[gid] + bvec[d];
    dt[gid] = (x > 20.0f) ? x : log1pf(__expf(x));
}

// ---------- selective scan + D skip + z gating; emits bf16 ----------
__global__ __launch_bounds__(256) void scan_kernel(const float* __restrict__ xdbl,
                                                   const float* __restrict__ dtb,
                                                   const float* __restrict__ xf,
                                                   const float* __restrict__ xz,
                                                   const float* __restrict__ Alog,
                                                   const float* __restrict__ Dp,
                                                   u16* __restrict__ yb) {
    const int n  = blockIdx.x >> 2;
    const int d  = ((blockIdx.x & 3) << 8) + threadIdx.x;
    const int tb = n * CHUNK;

    float Ad[S_DIM], st[S_DIM];
#pragma unroll
    for (int s = 0; s < S_DIM; s++) {
        Ad[s] = -__expf(Alog[(size_t)d * S_DIM + s]);
        st[s] = 0.0f;
    }
    const float Dd = Dp[d];

    __shared__ float sB[16 * S_DIM];
    __shared__ float sC[16 * S_DIM];

    for (int l0 = 0; l0 < CHUNK; l0 += 16) {
        int ns = CHUNK - l0; if (ns > 16) ns = 16;
        __syncthreads();
        for (int e = threadIdx.x; e < ns * S_DIM; e += 256) {
            int stp = e >> 4, s = e & 15;
            const float* row = xdbl + (size_t)(tb + l0 + stp) * (DT_RANK + 2 * S_DIM);
            sB[e] = row[DT_RANK + s];
            sC[e] = row[DT_RANK + S_DIM + s];
        }
        __syncthreads();
        for (int i = 0; i < ns; i++) {
            size_t t = (size_t)(tb + l0 + i);
            float dtv = dtb[t * DI + d];
            float xv  = xf[t * DI + d];
            float zv  = xz[t * (2 * DI) + DI + d];
            float dx  = dtv * xv;
            float y = 0.0f;
#pragma unroll
            for (int s = 0; s < S_DIM; s++) {
                st[s] = __expf(dtv * Ad[s]) * st[s] + dx * sB[i * S_DIM + s];
                y += st[s] * sC[i * S_DIM + s];
            }
            float out = (y + xv * Dd) * (zv * (1.0f / (1.0f + __expf(-zv))));
            yb[t * DI + d] = f2bf(out);
        }
    }
}

// ---------- host orchestration ----------
extern "C" void kernel_launch(void* const* d_in, const int* in_sizes, int n_in,
                              void* d_out, int out_size, void* d_ws, size_t ws_size,
                              hipStream_t stream) {
    const float* sf    = (const float*)d_in[0];
    const float* ip_w  = (const float*)d_in[1];
    const float* ip_b  = (const float*)d_in[2];
    const float* ilng  = (const float*)d_in[3];
    const float* ilnb  = (const float*)d_in[4];
    const float* m_inw = (const float*)d_in[5];
    const float* m_cw  = (const float*)d_in[6];
    const float* m_cb  = (const float*)d_in[7];
    const float* m_xpw = (const float*)d_in[8];
    const float* m_dtw = (const float*)d_in[9];
    const float* m_dtb = (const float*)d_in[10];
    const float* m_Al  = (const float*)d_in[11];
    const float* m_D   = (const float*)d_in[12];
    const float* m_ow  = (const float*)d_in[13];
    const float* lng   = (const float*)d_in[14];
    const float* lnb   = (const float*)d_in[15];
    const float* op_w  = (const float*)d_in[16];
    const float* op_b  = (const float*)d_in[17];
    float* out = (float*)d_out;

    size_t off = 0;
    auto carve = [&](size_t bytes) -> char* {
        char* p = (char*)d_ws + off;
        off += (bytes + 255) & ~(size_t)255;
        return p;
    };
    u16*   sfb   = (u16*)  carve((size_t)N_SLICES * IN_DIM * 2);
    u16*   wb_ip = (u16*)  carve((size_t)E_DIM * IN_DIM * 2);
    u16*   wb_in = (u16*)  carve((size_t)NL * 2 * DI * E_DIM * 2);
    u16*   wb_xp = (u16*)  carve((size_t)NL * (DT_RANK + 2 * S_DIM) * DI * 2);
    u16*   wb_dt = (u16*)  carve((size_t)NL * DI * DT_RANK * 2);
    u16*   wb_ou = (u16*)  carve((size_t)NL * E_DIM * DI * 2);
    u16*   wb_op = (u16*)  carve((size_t)E_DIM * E_DIM * 2);
    float* hf    = (float*)carve((size_t)N_SLICES * E_DIM * 4);
    u16*   hb    = (u16*)  carve((size_t)N_SLICES * E_DIM * 2);
    float* tmp   = (float*)carve((size_t)N_SLICES * E_DIM * 4);
    float* xzb   = (float*)carve((size_t)N_SLICES * 2 * DI * 4);
    float* xcf   = (float*)carve((size_t)N_SLICES * DI * 4);
    u16*   xcb   = (u16*)  carve((size_t)N_SLICES * DI * 2);
    float* xdbl  = (float*)carve((size_t)N_SLICES * (DT_RANK + 2 * S_DIM) * 4);
    u16*   dt32b = (u16*)  carve((size_t)N_SLICES * DT_RANK * 2);
    float* dtbuf = (float*)carve((size_t)N_SLICES * DI * 4);
    u16*   yb    = (u16*)  carve((size_t)N_SLICES * DI * 2);
    if (off > ws_size) return;

    auto cvt = [&](const float* s, u16* d, int n) {
        cvt_f32_bf16<<<(n + 255) / 256, 256, 0, stream>>>(s, d, n);
    };
    cvt(sf,    sfb,   N_SLICES * IN_DIM);
    cvt(ip_w,  wb_ip, E_DIM * IN_DIM);
    cvt(m_inw, wb_in, NL * 2 * DI * E_DIM);
    cvt(m_xpw, wb_xp, NL * (DT_RANK + 2 * S_DIM) * DI);
    cvt(m_dtw, wb_dt, NL * DI * DT_RANK);
    cvt(m_ow,  wb_ou, NL * E_DIM * DI);
    cvt(op_w,  wb_op, E_DIM * E_DIM);

    const int MB = N_SLICES / 64;  // 125 row blocks for the LDS GEMM

    // 1) input projection + LN
    gemm_bf16_wmma_lds<<<dim3(E_DIM / 256, MB), 256, 0, stream>>>(
        sfb, wb_ip, ip_b, tmp, N_SLICES, E_DIM, IN_DIM);
    ln_kernel<<<N_SLICES, 256, 0, stream>>>(tmp, nullptr, ilng, ilnb, hf, hb);

    // 2) Mamba layers
    for (int l = 0; l < NL; l++) {
        gemm_bf16_wmma_lds<<<dim3((2 * DI) / 256, MB), 256, 0, stream>>>(
            hb, wb_in + (size_t)l * 2 * DI * E_DIM, nullptr, xzb,
            N_SLICES, 2 * DI, E_DIM);
        conv_silu_kernel<<<(N_SLICES * DI + 255) / 256, 256, 0, stream>>>(
            xzb, m_cw + (size_t)l * DI * DC, m_cb + (size_t)l * DI, xcf, xcb);
        gemm_bf16_wmma_direct<1><<<dim3(1, N_SLICES / 16), 128, 0, stream>>>(
            xcb, wb_xp + (size_t)l * (DT_RANK + 2 * S_DIM) * DI, nullptr, xdbl,
            N_SLICES, DT_RANK + 2 * S_DIM, DI);
        cvt2d_f32_bf16<<<(N_SLICES * DT_RANK + 255) / 256, 256, 0, stream>>>(
            xdbl, dt32b, N_SLICES, DT_RANK + 2 * S_DIM, DT_RANK);
        gemm_bf16_wmma_lds<<<dim3(DI / 256, MB), 256, 0, stream>>>(
            dt32b, wb_dt + (size_t)l * DI * DT_RANK, nullptr, dtbuf,
            N_SLICES, DI, DT_RANK);
        softplus_kernel<<<(N_SLICES * DI + 255) / 256, 256, 0, stream>>>(
            dtbuf, m_dtb + (size_t)l * DI);
        scan_kernel<<<NCHUNK * (DI / 256), 256, 0, stream>>>(
            xdbl, dtbuf, xcf, xzb, m_Al + (size_t)l * DI * S_DIM,
            m_D + (size_t)l * DI, yb);
        gemm_bf16_wmma_lds<<<dim3(E_DIM / 256, MB), 256, 0, stream>>>(
            yb, wb_ou + (size_t)l * E_DIM * DI, nullptr, tmp,
            N_SLICES, E_DIM, DI);
        ln_kernel<<<N_SLICES, 256, 0, stream>>>(
            tmp, hf, lng + (size_t)l * E_DIM, lnb + (size_t)l * E_DIM, hf, hb);
    }

    // 3) output projection
    gemm_bf16_wmma_lds<<<dim3(E_DIM / 256, MB), 256, 0, stream>>>(
        hb, wb_op, op_b, out, N_SLICES, E_DIM, E_DIM);
}
